// DTransformer_79585743995625
// MI455X (gfx1250) — matrix operand order repaired and checked by reading, not compiled, not run on hardware
//
#include <hip/hip_runtime.h>
#include <hip/hip_bf16.h>
#include <math.h>

typedef __attribute__((ext_vector_type(16))) _Float16 v16h;
typedef __attribute__((ext_vector_type(8)))  float    v8f;
typedef __attribute__((ext_vector_type(4)))  float    f32x4;
typedef __attribute__((ext_vector_type(4)))  unsigned int u32x4;
typedef __attribute__((ext_vector_type(4)))  int      i32x4;
typedef __attribute__((ext_vector_type(8)))  int      i32x8;
typedef unsigned int u32;

#define Bb   2
#define Ss   512
#define Dd   256
#define Hh   8
#define DKk  32
#define NKk  16
#define NEGF (-1e32f)
#define INV_SQRT_DK 0.17677669529663687f

// ---------------------------------------------------------------------------
// WMMA helpers (wave32, V_WMMA_F32_16X16X32_F16; C/D layout per CDNA5 ISA 7.12.2, A/B use contiguous k = 16h + e)
// ---------------------------------------------------------------------------
__device__ inline v8f wmma_f16(v16h a, v16h b, v8f c) {
  return __builtin_amdgcn_wmma_f32_16x16x32_f16(
      /*neg_a=*/false, a, /*neg_b=*/false, b,
      /*c_mod=*/(short)0, c, /*reuse_a=*/false, /*reuse_b=*/false);
}

// A fragment 16x32 (MxK), f32 source row-major, 16B-aligned rows.
// Lanes 0-15: M=lane, halves 0..15 = K0..15.
// Lanes16-31: M=lane-16, halves 0..15 = K16..31.
__device__ inline v16h load_a16x32(const float* base, int ld, int lane) {
  int m = lane & 15;
  int koff = (lane >> 4) << 4;           // 0 or 16
  const float* rp = base + (size_t)m * ld + koff;
  f32x4 p0 = *(const f32x4*)(rp + 0);
  f32x4 p1 = *(const f32x4*)(rp + 4);
  f32x4 p2 = *(const f32x4*)(rp + 8);
  f32x4 p3 = *(const f32x4*)(rp + 12);
  v16h a;
#pragma unroll
  for (int e = 0; e < 4; ++e) a[e]      = (_Float16)p0[e];
#pragma unroll
  for (int e = 0; e < 4; ++e) a[4 + e]  = (_Float16)p1[e];
#pragma unroll
  for (int e = 0; e < 4; ++e) a[8 + e]  = (_Float16)p2[e];
#pragma unroll
  for (int e = 0; e < 4; ++e) a[12 + e] = (_Float16)p3[e];
  return a;
}

// B fragment 32x16 (KxN) where memory holds rows of K: B(k,n) = base[n*ld + k]
// (X @ W^T with W row-major [N,K], or Q @ K^T with K row-major [N,DK]).
__device__ inline v16h load_b_rowsK(const float* base, int ld, int lane) {
  int n = lane & 15;
  int kb = (lane >> 4) << 4;
  const float* rp = base + (size_t)n * ld + kb;
  f32x4 p0 = *(const f32x4*)(rp + 0);
  f32x4 p1 = *(const f32x4*)(rp + 4);
  f32x4 p2 = *(const f32x4*)(rp + 8);
  f32x4 p3 = *(const f32x4*)(rp + 12);
  v16h b;
#pragma unroll
  for (int e = 0; e < 4; ++e) b[e]      = (_Float16)p0[e];
#pragma unroll
  for (int e = 0; e < 4; ++e) b[4 + e]  = (_Float16)p1[e];
#pragma unroll
  for (int e = 0; e < 4; ++e) b[8 + e]  = (_Float16)p2[e];
#pragma unroll
  for (int e = 0; e < 4; ++e) b[12 + e] = (_Float16)p3[e];
  return b;
}

// B fragment 32x16 from an LDS-resident [Ss][DKk] V-head tile:
// B(k,n) = vtile[k0+kb+e][n0+n]
__device__ inline v16h load_b_lds(const float* base /*&vtile[k0][0]*/, int n0, int lane) {
  int n = (lane & 15) + n0;
  int kb = (lane >> 4) << 4;
  const float* cp = base + (size_t)kb * DKk + n;
  v16h b;
#pragma unroll
  for (int e = 0; e < 16; ++e) b[e] = (_Float16)cp[e * DKk];
  return b;
}

// C/D 16x16 f32: VGPR r holds (M=r, N=lane) lanes 0-15, (M=8+r, N=lane-16).
__device__ inline void store_tile(float* C, size_t ldc, v8f acc, int lane,
                                  float scale, float add) {
  int n = lane & 15;
  int mb = (lane >> 4) << 3;
#pragma unroll
  for (int r = 0; r < 8; ++r)
    C[(size_t)(mb + r) * ldc + n] = acc[r] * scale + add;
}

// ---------------------------------------------------------------------------
// TDM: stage a [512 rows x 32 cols] f32 column-window (row stride Dd) into
// LDS offset 0 via tensor_load_to_lds; fallback = vectorized copy loop.
// Issued by wave 0 only; caller must __syncthreads() after.
// ---------------------------------------------------------------------------
__device__ inline void stage_vhead(const float* g, float* lds, int t) {
#if __has_builtin(__builtin_amdgcn_tensor_load_to_lds)
  (void)lds;
  if (t < 32) {                           // one wave issues the DMA
    unsigned long long ga = (unsigned long long)(size_t)g;
    u32x4 g0;
    g0[0] = 1u;                           // count=1, user descriptor
    g0[1] = 0u;                           // lds_addr = 0 (vtile is sole LDS obj)
    g0[2] = (u32)ga;                      // global_addr[31:0]
    g0[3] = (u32)((ga >> 32) & 0x01FFFFFFull) | 0x80000000u; // addr[56:32]|type=2
    i32x8 g1;
    g1[0] = (int)(2u << 16);              // data_size = 4 bytes
    g1[1] = (int)((u32)DKk << 16);        // tensor_dim0 = 32 (bits 79:48 lo)
    g1[2] = (int)((u32)Ss  << 16);        // tensor_dim1 = 512 (bits 111:80 lo)
    g1[3] = (int)((u32)DKk << 16);        // tile_dim0 = 32 (bits 127:112)
    g1[4] = (int)Ss;                      // tile_dim1 = 512 (bits 143:128)
    g1[5] = (int)Dd;                      // tensor_dim0_stride = 256 (bits 207:160)
    g1[6] = 0;
    g1[7] = 0;
    i32x4 z4 = {0, 0, 0, 0};
#if __clang_major__ >= 23
    i32x8 z8 = {0, 0, 0, 0, 0, 0, 0, 0};
    __builtin_amdgcn_tensor_load_to_lds(g0, g1, z4, z4, z8, 0);
#else
    __builtin_amdgcn_tensor_load_to_lds(g0, g1, z4, z4, 0);
#endif
    __builtin_amdgcn_s_wait_tensorcnt(0);
  }
#else
  for (int i = t; i < (Ss * DKk) / 4; i += 64) {
    int row = i >> 3, c4 = i & 7;
    ((f32x4*)lds)[i] = *(const f32x4*)(g + (size_t)row * Dd + c4 * 4);
  }
#endif
}

// ---------------------------------------------------------------------------
// Block reductions (blockDim.x == 256)
// ---------------------------------------------------------------------------
__device__ inline float blk_max(float v, float* red) {
  int t = threadIdx.x;
  red[t] = v; __syncthreads();
#pragma unroll
  for (int o = 128; o > 0; o >>= 1) {
    if (t < o) red[t] = fmaxf(red[t], red[t + o]);
    __syncthreads();
  }
  float r = red[0]; __syncthreads();
  return r;
}

__device__ inline float blk_sum(float v, float* red) {
  int t = threadIdx.x;
  red[t] = v; __syncthreads();
#pragma unroll
  for (int o = 128; o > 0; o >>= 1) {
    if (t < o) red[t] += red[t + o];
    __syncthreads();
  }
  float r = red[0]; __syncthreads();
  return r;
}

// ---------------------------------------------------------------------------
// Generic GEMM: C[M,N] = A[M,K] @ W[N,K]^T + bias[N]   (f32 io, f16 WMMA)
// block = 128 (4 waves), wave w -> N-tile blockIdx.y*4+w, M-tile = blockIdx.x
// ---------------------------------------------------------------------------
__global__ __launch_bounds__(128)
void gemm_xwT_bias_kernel(const float* __restrict__ A, const float* __restrict__ W,
                          const float* __restrict__ bias, float* __restrict__ C,
                          int M, int N, int K) {
  int lane = threadIdx.x & 31;
  int wave = threadIdx.x >> 5;
  int mt = blockIdx.x;
  int nt = blockIdx.y * 4 + wave;
  const float* Abase = A + (size_t)mt * 16 * K;
  const float* Wbase = W + (size_t)nt * 16 * K;
  v8f acc = {};
  for (int k0 = 0; k0 < K; k0 += 32) {
    if (k0 + 32 < K)  // global_prefetch_b8 hint for next A tile
      __builtin_prefetch(Abase + (size_t)(lane & 15) * K + k0 + 32, 0, 1);
    v16h a = load_a16x32(Abase + k0, K, lane);
    v16h b = load_b_rowsK(Wbase + k0, K, lane);
    acc = wmma_f16(a, b, acc);
  }
  float add = bias[nt * 16 + (lane & 15)];
  store_tile(C + (size_t)mt * 16 * N + nt * 16, (size_t)N, acc, lane, 1.0f, add);
}

// ---------------------------------------------------------------------------
// Block1 raw scores: sraw[b,h,q,k] = (q1_h q · q1_h k)/sqrt(DK)   (kq_same)
// grid(S/16, S/16/8, B*H), block 256 (8 waves along k-tiles). K = DK = 32.
// ---------------------------------------------------------------------------
__global__ __launch_bounds__(256)
void attn1_scores_kernel(const float* __restrict__ q1, float* __restrict__ sraw) {
  int lane = threadIdx.x & 31;
  int wave = threadIdx.x >> 5;
  int qt = blockIdx.x;
  int kt = blockIdx.y * 8 + wave;
  int b = blockIdx.z / Hh, h = blockIdx.z % Hh;
  const float* head = q1 + (size_t)b * Ss * Dd + h * DKk;
  v16h a  = load_a16x32(head + (size_t)qt * 16 * Dd, Dd, lane);
  v16h bf = load_b_rowsK(head + (size_t)kt * 16 * Dd, Dd, lane);
  v8f acc = {};
  acc = wmma_f16(a, bf, acc);
  float* out = sraw + ((size_t)blockIdx.z * Ss + (size_t)qt * 16) * Ss + kt * 16;
  store_tile(out, (size_t)Ss, acc, lane, INV_SQRT_DK, 0.0f);
}

// ---------------------------------------------------------------------------
// Block1 row pipeline: masked softmax -> cumsum -> decay -> masked softmax.
// One block per (b,h,q); 256 threads each own k and k+256. Writes q_scores.
// ---------------------------------------------------------------------------
__global__ __launch_bounds__(256)
void attn1_rows_kernel(const float* __restrict__ sraw, const float* __restrict__ gam,
                       float* __restrict__ qsc) {
  __shared__ float sb[2][Ss];
  __shared__ float red[256];
  int t = threadIdx.x;
  size_t row = blockIdx.x;                 // (b*H + h)*S + q
  int q = (int)(row % Ss);
  int h = (int)((row / Ss) % Hh);
  const float* rp = sraw + row * Ss;
  int k0i = t, k1i = t + 256;
  float r0 = rp[k0i], r1 = rp[k1i];
  bool v0 = (k0i <= q), v1 = (k1i <= q);   // peek_cur=True mask
  float x0 = v0 ? r0 : NEGF, x1 = v1 ? r1 : NEGF;
  float mx = blk_max(fmaxf(x0, x1), red);
  float e0 = expf(x0 - mx), e1 = expf(x1 - mx);
  float inv = 1.0f / blk_sum(e0 + e1, red);
  sb[0][k0i] = e0 * inv; sb[0][k1i] = e1 * inv;
  __syncthreads();
  int cur = 0;
  for (int off = 1; off < Ss; off <<= 1) { // inclusive scan (cumsum of softmax)
    int nxt = cur ^ 1;
    sb[nxt][k0i] = sb[cur][k0i] + (k0i >= off ? sb[cur][k0i - off] : 0.0f);
    sb[nxt][k1i] = sb[cur][k1i] + (k1i >= off ? sb[cur][k1i - off] : 0.0f);
    __syncthreads();
    cur = nxt;
  }
  float c0 = sb[cur][k0i], c1 = sb[cur][k1i];
  float g = -fabsf(gam[h]);
  float d0 = sqrtf(fmaxf((1.0f - c0) * fabsf((float)(q - k0i)), 0.0f));
  float d1 = sqrtf(fmaxf((1.0f - c1) * fabsf((float)(q - k1i)), 0.0f));
  float te0 = fminf(fmaxf(expf(d0 * g), 1e-5f), 1e5f);
  float te1 = fminf(fmaxf(expf(d1 * g), 1e-5f), 1e5f);
  float y0 = v0 ? r0 * te0 : NEGF, y1 = v1 ? r1 * te1 : NEGF;
  float mx2 = blk_max(fmaxf(y0, y1), red);
  float f0 = expf(y0 - mx2), f1 = expf(y1 - mx2);
  float inv2 = 1.0f / blk_sum(f0 + f1, red);
  float* op = qsc + row * Ss;
  op[k0i] = v0 ? f0 * inv2 : 0.0f;
  op[k1i] = v1 ? f1 * inv2 : 0.0f;
}

// ---------------------------------------------------------------------------
// Block1 scores @ V with TDM-staged V head in LDS.
// grid(4, 1, B*H), block 64 (2 waves = two 16-wide N tiles of DK=32);
// each block loops 8 q-tiles, amortizing one tensor_load_to_lds of [512x32].
// ---------------------------------------------------------------------------
__global__ __launch_bounds__(64)
void attn1_av_kernel(const float* __restrict__ qsc, const float* __restrict__ v1,
                     float* __restrict__ concat1) {
  __shared__ float vtile[Ss][DKk];         // 64 KB, sole LDS object (offset 0)
  int t = threadIdx.x;
  int lane = t & 31;
  int wave = t >> 5;
  int b = blockIdx.z / Hh, h = blockIdx.z % Hh;
  stage_vhead(v1 + (size_t)b * Ss * Dd + h * DKk, &vtile[0][0], t);
  __syncthreads();
  for (int qi = 0; qi < 8; ++qi) {
    int qt = blockIdx.x * 8 + qi;
    const float* A = qsc + ((size_t)blockIdx.z * Ss + (size_t)qt * 16) * Ss;
    v8f acc = {};
    for (int k0 = 0; k0 < Ss; k0 += 32) {
      v16h a  = load_a16x32(A + k0, Ss, lane);
      v16h bf = load_b_lds(&vtile[k0][0], wave * 16, lane);
      acc = wmma_f16(a, bf, acc);
    }
    float* out = concat1 + ((size_t)b * Ss + (size_t)qt * 16) * Dd + h * DKk + wave * 16;
    store_tile(out, (size_t)Dd, acc, lane, 1.0f, 0.0f);
  }
}

// ---------------------------------------------------------------------------
// Residual + LayerNorm: out_row = LN(x_row + y_row) * g + b.  1 row / block.
// ---------------------------------------------------------------------------
__global__ __launch_bounds__(256)
void add_ln_kernel(const float* __restrict__ x, const float* __restrict__ y,
                   const float* __restrict__ g, const float* __restrict__ bv,
                   float* __restrict__ out) {
  __shared__ float red[256];
  int t = threadIdx.x;
  size_t row = blockIdx.x;
  float v = x[row * Dd + t] + y[row * Dd + t];
  float mean = blk_sum(v, red) * (1.0f / Dd);
  float d = v - mean;
  float var = blk_sum(d * d, red) * (1.0f / Dd);
  out[row * Dd + t] = d * rsqrtf(var + 1e-5f) * g[t] + bv[t];
}

// ---------------------------------------------------------------------------
// Block4 raw score vectors: s0[b,kq,h,kpos] (queries constant along s).
// ---------------------------------------------------------------------------
__global__ __launch_bounds__(256)
void attn4_s0_kernel(const float* __restrict__ q4k, const float* __restrict__ k4,
                     float* __restrict__ s0) {
  int idx = blockIdx.x * 256 + threadIdx.x;      // [B,NK,H,S] flat
  int kp = idx % Ss;
  int h  = (idx / Ss) % Hh;
  int kq = (idx / (Ss * Hh)) % NKk;
  int b  = idx / (Ss * Hh * NKk);
  const float* qp = q4k + kq * Dd + h * DKk;
  const float* kv = k4 + ((size_t)b * Ss + kp) * Dd + h * DKk;
  float acc = 0.0f;
#pragma unroll
  for (int d = 0; d < DKk; ++d) acc = fmaf(qp[d], kv[d], acc);
  s0[idx] = acc * INV_SQRT_DK;
}

// ---------------------------------------------------------------------------
// Block4 row pipeline (strict mask, maxout=True), writes k_scores directly
// in the final permuted layout [B,H,S,NK,S]. One block per (b,kq,h,q).
// ---------------------------------------------------------------------------
__global__ __launch_bounds__(256)
void attn4_rows_kernel(const float* __restrict__ s0, const float* __restrict__ gam,
                       float* __restrict__ ksc) {
  __shared__ float sb[2][Ss];
  __shared__ float red[256];
  int t = threadIdx.x;
  size_t idx = blockIdx.x;                 // ((b*NK+kq)*H + h)*S + q
  int q  = (int)(idx % Ss);
  int h  = (int)((idx / Ss) % Hh);
  int kq = (int)((idx / ((size_t)Ss * Hh)) % NKk);
  int b  = (int)(idx / ((size_t)Ss * Hh * NKk));
  const float* rp = s0 + (((size_t)b * NKk + kq) * Hh + h) * Ss;
  int k0i = t, k1i = t + 256;
  float r0 = rp[k0i], r1 = rp[k1i];
  bool v0 = (k0i < q), v1 = (k1i < q);     // strict causal mask
  float x0 = v0 ? r0 : NEGF, x1 = v1 ? r1 : NEGF;
  float mx = blk_max(fmaxf(x0, x1), red);
  float e0 = expf(x0 - mx), e1 = expf(x1 - mx);
  float inv = 1.0f / blk_sum(e0 + e1, red);
  sb[0][k0i] = e0 * inv; sb[0][k1i] = e1 * inv;
  __syncthreads();
  int cur = 0;
  for (int off = 1; off < Ss; off <<= 1) {
    int nxt = cur ^ 1;
    sb[nxt][k0i] = sb[cur][k0i] + (k0i >= off ? sb[cur][k0i - off] : 0.0f);
    sb[nxt][k1i] = sb[cur][k1i] + (k1i >= off ? sb[cur][k1i - off] : 0.0f);
    __syncthreads();
    cur = nxt;
  }
  float c0 = sb[cur][k0i], c1 = sb[cur][k1i];
  float g = -fabsf(gam[h]);
  float d0 = sqrtf(fmaxf((1.0f - c0) * fabsf((float)(q - k0i)), 0.0f));
  float d1 = sqrtf(fmaxf((1.0f - c1) * fabsf((float)(q - k1i)), 0.0f));
  float te0 = fminf(fmaxf(expf(d0 * g), 1e-5f), 1e5f);
  float te1 = fminf(fmaxf(expf(d1 * g), 1e-5f), 1e5f);
  float y0 = v0 ? r0 * te0 : NEGF, y1 = v1 ? r1 * te1 : NEGF;
  float mx2 = blk_max(fmaxf(y0, y1), red);
  float f0 = expf(y0 - mx2), f1 = expf(y1 - mx2);
  float inv2 = 1.0f / blk_sum(f0 + f1, red);
  float o0 = v0 ? f0 * inv2 : 0.0f;
  float o1 = v1 ? f1 * inv2 : 0.0f;
  float mxs = blk_max(fmaxf(o0, o1), red); // maxout
  float sc = fminf(1.0f / mxs, 5.0f);      // 1/0 -> inf -> clamp 5
  float* op = ksc + (((size_t)(b * Hh + h) * Ss + q) * NKk + kq) * Ss;
  op[k0i] = o0 * sc;
  op[k1i] = o1 * sc;
}

// ---------------------------------------------------------------------------
// Block4 scores @ V with TDM-staged V head in LDS; A tiles streamed from the
// permuted k_scores (row stride NK*S) as b128 loads.
// grid(4, 1, B*NK*H), block 64; each block loops 8 q-tiles.
// ---------------------------------------------------------------------------
__global__ __launch_bounds__(64)
void attn4_av_kernel(const float* __restrict__ ksc, const float* __restrict__ v4,
                     float* __restrict__ concat4) {
  __shared__ float vtile[Ss][DKk];         // 64 KB, sole LDS object (offset 0)
  int t = threadIdx.x;
  int lane = t & 31;
  int wave = t >> 5;
  int z = blockIdx.z;                      // (b*NK + kq)*H + h
  int h = z % Hh;
  int bk = z / Hh;
  int kq = bk % NKk;
  int b = bk / NKk;
  stage_vhead(v4 + (size_t)b * Ss * Dd + h * DKk, &vtile[0][0], t);
  __syncthreads();
  const int lda = NKk * Ss;                // 8192
  for (int qi = 0; qi < 8; ++qi) {
    int qt = blockIdx.x * 8 + qi;
    const float* A = ksc + (((size_t)(b * Hh + h) * Ss + (size_t)qt * 16) * NKk + kq) * Ss;
    v8f acc = {};
    for (int k0 = 0; k0 < Ss; k0 += 32) {
      v16h a  = load_a16x32(A + k0, lda, lane);
      v16h bf = load_b_lds(&vtile[k0][0], wave * 16, lane);
      acc = wmma_f16(a, bf, acc);
    }
    float* out = concat4 + ((size_t)bk * Ss + (size_t)qt * 16) * Dd + h * DKk + wave * 16;
    store_tile(out, (size_t)Dd, acc, lane, 1.0f, 0.0f);
  }
}

// ---------------------------------------------------------------------------
// Final residual (know_params) + LN, with the [B,NK,S,D]->[B,S,NK*D] permute.
// ---------------------------------------------------------------------------
__global__ __launch_bounds__(256)
void z_ln_kernel(const float* __restrict__ know, const float* __restrict__ zlin,
                 const float* __restrict__ g, const float* __restrict__ bv,
                 float* __restrict__ zout) {
  __shared__ float red[256];
  int t = threadIdx.x;
  size_t row = blockIdx.x;                 // (b*NK + kq)*S + s
  int s  = (int)(row % Ss);
  int kq = (int)((row / Ss) % NKk);
  int b  = (int)(row / ((size_t)Ss * NKk));
  float v = know[kq * Dd + t] + zlin[row * Dd + t];
  float mean = blk_sum(v, red) * (1.0f / Dd);
  float d = v - mean;
  float var = blk_sum(d * d, red) * (1.0f / Dd);
  zout[((size_t)(b * Ss + s) * NKk + kq) * Dd + t] =
      d * rsqrtf(var + 1e-5f) * g[t] + bv[t];
}

// ---------------------------------------------------------------------------
extern "C" void kernel_launch(void* const* d_in, const int* in_sizes, int n_in,
                              void* d_out, int out_size, void* d_ws, size_t ws_size,
                              hipStream_t stream) {
  (void)in_sizes; (void)n_in; (void)out_size; (void)ws_size;
  const float* q_emb = (const float*)d_in[0];
  const float* s_emb = (const float*)d_in[1];
  const float* Wq1 = (const float*)d_in[5];  const float* bq1 = (const float*)d_in[6];
  const float* Wv1 = (const float*)d_in[7];  const float* bv1 = (const float*)d_in[8];
  const float* Wo1 = (const float*)d_in[9];  const float* bo1 = (const float*)d_in[10];
  const float* gam1 = (const float*)d_in[11];
  const float* ln1g = (const float*)d_in[12]; const float* ln1b = (const float*)d_in[13];
  const float* Wq4 = (const float*)d_in[14]; const float* bq4 = (const float*)d_in[15];
  const float* Wk4 = (const float*)d_in[16]; const float* bk4 = (const float*)d_in[17];
  const float* Wv4 = (const float*)d_in[18]; const float* bv4 = (const float*)d_in[19];
  const float* Wo4 = (const float*)d_in[20]; const float* bo4 = (const float*)d_in[21];
  const float* gam4 = (const float*)d_in[22];
  const float* ln4g = (const float*)d_in[23]; const float* ln4b = (const float*)d_in[24];
  const float* know = (const float*)d_in[25];

  // workspace layout (floats)
  float* ws  = (float*)d_ws;
  float* q1  = ws;                 // B*S*D      = 262144
  float* v1  = q1  + 262144;
  float* sraw= v1  + 262144;       // B*H*S*S    = 4194304
  float* cc1 = sraw+ 4194304;      // 262144
  float* t1  = cc1 + 262144;       // 262144
  float* p   = t1  + 262144;       // 262144
  float* k4  = p   + 262144;       // 262144
  float* v4  = k4  + 262144;       // 262144
  float* q4k = v4  + 262144;       // NK*D       = 4096
  float* s0  = q4k + 4096;         // B*NK*H*S   = 131072
  float* cc4 = s0  + 131072;       // B*NK*S*D   = 4194304
  float* zl  = cc4 + 4194304;      // 4194304

  // outputs: z | q_scores | k_scores
  float* zout = (float*)d_out;
  float* qsc  = zout + 4194304;    // [B,H,S,S]
  float* ksc  = qsc  + 4194304;    // [B,H,S,NK,S]

  const dim3 gblk(128);
  // ---- block 1 ----
  gemm_xwT_bias_kernel<<<dim3(64, 4), gblk, 0, stream>>>(q_emb, Wq1, bq1, q1, 1024, 256, 256);
  gemm_xwT_bias_kernel<<<dim3(64, 4), gblk, 0, stream>>>(s_emb, Wv1, bv1, v1, 1024, 256, 256);
  attn1_scores_kernel<<<dim3(32, 4, Bb * Hh), 256, 0, stream>>>(q1, sraw);
  attn1_rows_kernel<<<dim3(Bb * Hh * Ss), 256, 0, stream>>>(sraw, gam1, qsc);
  attn1_av_kernel<<<dim3(4, 1, Bb * Hh), 64, 0, stream>>>(qsc, v1, cc1);
  gemm_xwT_bias_kernel<<<dim3(64, 4), gblk, 0, stream>>>(cc1, Wo1, bo1, t1, 1024, 256, 256);
  add_ln_kernel<<<dim3(Bb * Ss), 256, 0, stream>>>(q_emb, t1, ln1g, ln1b, p);
  // ---- block 4 projections ----
  gemm_xwT_bias_kernel<<<dim3(1, 4), gblk, 0, stream>>>(know, Wq4, bq4, q4k, 16, 256, 256);
  gemm_xwT_bias_kernel<<<dim3(64, 4), gblk, 0, stream>>>(q_emb, Wk4, bk4, k4, 1024, 256, 256);
  gemm_xwT_bias_kernel<<<dim3(64, 4), gblk, 0, stream>>>(p, Wv4, bv4, v4, 1024, 256, 256);
  // ---- block 4 attention ----
  attn4_s0_kernel<<<dim3((Bb * NKk * Hh * Ss) / 256), 256, 0, stream>>>(q4k, k4, s0);
  attn4_rows_kernel<<<dim3(Bb * NKk * Hh * Ss), 256, 0, stream>>>(s0, gam4, ksc);
  attn4_av_kernel<<<dim3(4, 1, Bb * NKk * Hh), 64, 0, stream>>>(ksc, v4, cc4);
  gemm_xwT_bias_kernel<<<dim3(1024, 4), gblk, 0, stream>>>(cc4, Wo4, bo4, zl, 16384, 256, 256);
  z_ln_kernel<<<dim3(Bb * NKk * Ss), 256, 0, stream>>>(know, zl, ln4g, ln4b, zout);
}
